// MultiHeadAttention_61924838473802
// MI455X (gfx1250) — compile-verified, matrix operand order checked by reading, not run on hardware
//
#include <hip/hip_runtime.h>

typedef __attribute__((ext_vector_type(16))) _Float16 v16h;
typedef __attribute__((ext_vector_type(8)))  _Float16 v8h;
typedef __attribute__((ext_vector_type(8)))  float    v8f;
typedef __attribute__((ext_vector_type(4)))  unsigned int u32x4;
typedef __attribute__((ext_vector_type(8)))  int i32x8;
typedef __attribute__((ext_vector_type(4)))  int i32x4;

#define WMMA_F16(a, b, c) \
    __builtin_amdgcn_wmma_f32_16x16x32_f16(false, (a), false, (b), (short)0, (c), false, false)

constexpr int Bz = 8, Nz = 2048, Cz = 768, Hz = 12, Dz = 64;
constexpr int Mtot = Bz * Nz;   // 16384 tokens
constexpr int F3   = 3 * Cz;    // 2304 qkv outputs

// ---------------------------------------------------------------------------
// helpers
// ---------------------------------------------------------------------------
__device__ inline v16h cvt16(float4 f0, float4 f1, float4 f2, float4 f3) {
    v16h r;
    r[0]=(_Float16)f0.x;  r[1]=(_Float16)f0.y;  r[2]=(_Float16)f0.z;  r[3]=(_Float16)f0.w;
    r[4]=(_Float16)f1.x;  r[5]=(_Float16)f1.y;  r[6]=(_Float16)f1.z;  r[7]=(_Float16)f1.w;
    r[8]=(_Float16)f2.x;  r[9]=(_Float16)f2.y;  r[10]=(_Float16)f2.z; r[11]=(_Float16)f2.w;
    r[12]=(_Float16)f3.x; r[13]=(_Float16)f3.y; r[14]=(_Float16)f3.z; r[15]=(_Float16)f3.w;
    return r;
}
__device__ inline v16h cat8(v8h lo, v8h hi) {
    return __builtin_shufflevector(lo, hi, 0,1,2,3,4,5,6,7,8,9,10,11,12,13,14,15);
}

struct Raw4 { float4 v0, v1, v2, v3; };
__device__ inline Raw4 load_raw(const float* __restrict__ row, int kb, int g) {
    const float4* p = (const float4*)(row + kb);
    Raw4 r; r.v0 = p[2*g]; r.v1 = p[2*g + 1]; r.v2 = p[4 + 2*g]; r.v3 = p[5 + 2*g];
    return r;
}
__device__ inline v16h cvt_raw(const Raw4& r) { return cvt16(r.v0, r.v1, r.v2, r.v3); }

struct RawH { v8h lo, hi; };
__device__ inline RawH load_rawh(const v8h* __restrict__ row, int c0, int g) {
    RawH r; r.lo = row[c0 + g]; r.hi = row[c0 + 2 + g]; return r;
}

// LDS byte offset of a __shared__ object (addrspace(3) ptrtoint).
__device__ inline unsigned lds_off(const void* p) {
    return (unsigned)(unsigned long long)(const __attribute__((address_space(3))) void*)p;
}

// Issue one 2-D TDM tile load (Tensor Data Mover) into LDS.
//   data_size_code: 1 = 2B elements, 2 = 4B elements
//   pad: (pad_amount_code+1) DWORDs inserted every 2<<pad_interval_code DWORDs
__device__ __attribute__((always_inline)) inline
void tdm_load_2d(unsigned lds_byte, const void* gaddr, unsigned data_size_code,
                 unsigned tensor_d0, unsigned tensor_d1,
                 unsigned tile_d0, unsigned tile_d1, unsigned stride0,
                 unsigned pad_interval_code, unsigned pad_amount_code)
{
    const unsigned long long ga = (unsigned long long)gaddr;
    u32x4 g0;
    g0[0] = 1u;                                              // count=1, user desc
    g0[1] = lds_byte;                                        // lds_addr
    g0[2] = (unsigned)(ga & 0xFFFFFFFFu);                    // global_addr[31:0]
    g0[3] = (unsigned)((ga >> 32) & 0x1FFFFFFu) | (2u << 30);// addr[56:32] | type=2
    i32x8 g1;
    g1[0] = (int)((data_size_code << 16) | (1u << 20) |      // pad_enable
                  (pad_interval_code << 22) | (pad_amount_code << 25));
    g1[1] = (int)((tensor_d0 & 0xFFFFu) << 16);              // dim0[15:0] @ bits 63:48
    g1[2] = (int)(((tensor_d0 >> 16) & 0xFFFFu) | ((tensor_d1 & 0xFFFFu) << 16));
    g1[3] = (int)(((tensor_d1 >> 16) & 0xFFFFu) | (tile_d0 << 16));
    g1[4] = (int)(tile_d1 & 0xFFFFu);                        // tile_dim2 = 0 (2D)
    g1[5] = (int)stride0;                                    // dim0_stride[31:0]
    g1[6] = 0;
    g1[7] = 0;
    const i32x4 gz = {0, 0, 0, 0};
#if __has_include(<hip/amd_detail/amd_gfx1250_TDM.h>)
    const i32x8 gz8 = {0, 0, 0, 0, 0, 0, 0, 0};
    __builtin_amdgcn_tensor_load_to_lds(g0, g1, gz, gz, gz8, 0);
#else
    __builtin_amdgcn_tensor_load_to_lds(g0, g1, gz, gz, 0);
#endif
}

// ---------------------------------------------------------------------------
// Kernel 1: qkv = x @ w_qkv^T  (fp32 in, f16 out to workspace)
// Block = 4 waves sharing one 64-row weight panel; weights are TDM-staged
// into double-buffered LDS (padded rows: 36 floats -> conflict-free b128).
// Always-issue wrap-around prefetch keeps s_wait_tensorcnt a literal 1.
// ---------------------------------------------------------------------------
__global__ __launch_bounds__(128)
void qkv_kernel(const float* __restrict__ x, const float* __restrict__ w,
                _Float16* __restrict__ qb, _Float16* __restrict__ kbuf,
                _Float16* __restrict__ vtb)
{
    __shared__ alignas(16) float wstage[2][64 * 36];   // 2 x 9216 B

    const int wid  = threadIdx.x >> 5;
    const int lane = threadIdx.x & 31;
    const int g    = lane >> 4;
    const int n    = lane & 15;

    const int tile  = blockIdx.x * 4 + wid;
    const int TM    = Mtot / 16;             // 1024 (all 4 waves share tf)
    const int tm    = tile % TM;
    const int tf    = tile / TM;             // 0..35
    const int mbase = tm * 16;
    const int fbase = tf * 64;

    v8f acc[4] = {};
    const float* xrow  = x + (size_t)(mbase + n) * Cz;
    const float* wbase = w + (size_t)fbase * Cz;      // step s tile at +s*32 floats

    constexpr int NS = Cz / 32;   // 24
    Raw4 ra[2];
    ra[0] = load_raw(xrow, 0, g);
    if (wid == 0)
        tdm_load_2d(lds_off(&wstage[0][0]), wbase, 2, Cz, F3, 32, 64, Cz, 4, 3);

#pragma unroll
    for (int s = 0; s < NS; ++s) {
        const int cur = s & 1, nxt = cur ^ 1;
        const int snext = (s + 1 < NS) ? (s + 1) : 0;   // wrap: last prefetch is dead
        ra[nxt] = load_raw(xrow, snext * 32, g);
        if (wid == 0) {
            tdm_load_2d(lds_off(&wstage[nxt][0]), wbase + snext * 32,
                        2, Cz, F3, 32, 64, Cz, 4, 3);
            __builtin_amdgcn_s_wait_tensorcnt(1);       // tile s complete
        }
        __syncthreads();

        const v16h av = cvt_raw(ra[cur]);
#pragma unroll
        for (int t = 0; t < 4; ++t) {
            const float4* lrow = (const float4*)&wstage[cur][(16 * t + n) * 36];
            const v16h bv = cvt16(lrow[2*g], lrow[2*g + 1], lrow[4 + 2*g], lrow[5 + 2*g]);
            acc[t] = WMMA_F16(av, bv, acc[t]);
        }
        __syncthreads();
    }

#pragma unroll
    for (int t = 0; t < 4; ++t) {
        const int f = fbase + 16*t + n;
#pragma unroll
        for (int r = 0; r < 8; ++r) {
            const int tok  = mbase + r + 8*g;
            const int bb   = tok >> 11;
            const int nseq = tok & 2047;
            const float v  = acc[t][r];
            if (f < Cz) {                                           // Q (scaled)
                const int h = f >> 6, d = f & 63;
                qb[(((size_t)bb * Hz + h) * Nz + nseq) * Dz + d] = (_Float16)(v * 0.125f);
            } else if (f < 2 * Cz) {                                // K
                const int f2 = f - Cz, h = f2 >> 6, d = f2 & 63;
                kbuf[(((size_t)bb * Hz + h) * Nz + nseq) * Dz + d] = (_Float16)v;
            } else {                                                // V transposed
                const int f2 = f - 2 * Cz, h = f2 >> 6, d = f2 & 63;
                vtb[(((size_t)bb * Hz + h) * Dz + d) * Nz + nseq] = (_Float16)v;
            }
        }
    }
}

// ---------------------------------------------------------------------------
// Kernel 2: flash attention. Block = 4 waves sharing one (b,h); K and V
// 32-key tiles are TDM-staged into double-buffered LDS (always-issue
// wrap-around prefetch -> constant s_wait_tensorcnt 2).
// ---------------------------------------------------------------------------
__global__ __launch_bounds__(128)
void attn_kernel(const _Float16* __restrict__ qb, const _Float16* __restrict__ kbuf,
                 const _Float16* __restrict__ vtb, _Float16* __restrict__ attn)
{
    __shared__ alignas(16) _Float16 kst[2][32 * 72];   // 32 keys x (64+8) halves
    __shared__ alignas(16) _Float16 vst[2][64 * 40];   // 64 dims x (32+8) halves
    __shared__ alignas(16) _Float16 plds[4][16][32];   // per-wave P staging

    const int wid  = threadIdx.x >> 5;
    const int lane = threadIdx.x & 31;
    const int g    = lane >> 4;
    const int n    = lane & 15;

    const int tile  = blockIdx.x * 4 + wid;
    const int mt    = tile & 127;     // all 4 waves share bh
    const int bh    = tile >> 7;      // 0..95
    const int mbase = mt * 16;

    const _Float16* qp = qb   + (size_t)bh * Nz * Dz;
    const _Float16* kp = kbuf + (size_t)bh * Nz * Dz;
    const _Float16* vp = vtb  + (size_t)bh * Dz * Nz;

    // Q tile (16 rows x 64 d) -> two v16h per lane
    const v8h* qrow = (const v8h*)(qp + (size_t)(mbase + n) * Dz);
    const v16h qa0 = cat8(qrow[0 + g], qrow[2 + g]);
    const v16h qa1 = cat8(qrow[4 + g], qrow[6 + g]);

    v8f O0 = {}, O1 = {}, O2 = {}, O3 = {};
    float m[8], l[8];
#pragma unroll
    for (int r = 0; r < 8; ++r) { m[r] = -3.0e38f; l[r] = 0.f; }

    if (wid == 0) {   // prologue: stage step 0
        tdm_load_2d(lds_off(&kst[0][0]), kp, 1, Dz, Nz, 64, 32, Dz, 4, 3);
        tdm_load_2d(lds_off(&vst[0][0]), vp, 1, Nz, Dz, 32, 64, Nz, 3, 3);
    }

    constexpr int NSK = Nz / 32;   // 64 key steps
    for (int s = 0; s < NSK; ++s) {
        const int cur = s & 1, nxt = cur ^ 1;
        if (wid == 0) {   // always-issue (wrap) so the wait count is constant
            const int kn = ((s + 1) & (NSK - 1)) * 32;
            tdm_load_2d(lds_off(&kst[nxt][0]), kp + (size_t)kn * Dz, 1, Dz, Nz, 64, 32, Dz, 4, 3);
            tdm_load_2d(lds_off(&vst[nxt][0]), vp + (size_t)kn,      1, Nz, Dz, 32, 64, Nz, 3, 3);
            __builtin_amdgcn_s_wait_tensorcnt(2);   // step-s pair complete
        }
        __syncthreads();

        // ---- K operand from LDS (rows n / 16+n, 144-B padded stride) -------
        const v8h* kr0 = (const v8h*)&kst[cur][(unsigned)n * 72];
        const v8h* kr1 = (const v8h*)&kst[cur][(unsigned)(16 + n) * 72];
        const v8h k00 = kr0[0 + g], k01 = kr0[2 + g], k02 = kr0[4 + g], k03 = kr0[6 + g];
        const v8h k10 = kr1[0 + g], k11 = kr1[2 + g], k12 = kr1[4 + g], k13 = kr1[6 + g];

        v8f S0 = {}, S1 = {};
        S0 = WMMA_F16(qa0, cat8(k00, k01), S0);
        S0 = WMMA_F16(qa1, cat8(k02, k03), S0);
        S1 = WMMA_F16(qa0, cat8(k10, k11), S1);
        S1 = WMMA_F16(qa1, cat8(k12, k13), S1);

        // ---- online softmax -------------------------------------------------
        float p0[8], p1[8], corr[8];
#pragma unroll
        for (int r = 0; r < 8; ++r) {
            float rm = fmaxf(S0[r], S1[r]);
            rm = fmaxf(rm, __shfl_xor(rm, 1, 32));
            rm = fmaxf(rm, __shfl_xor(rm, 2, 32));
            rm = fmaxf(rm, __shfl_xor(rm, 4, 32));
            rm = fmaxf(rm, __shfl_xor(rm, 8, 32));
            const float mn = fmaxf(m[r], rm);
            corr[r] = __expf(m[r] - mn);
            p0[r]   = __expf(S0[r] - mn);
            p1[r]   = __expf(S1[r] - mn);
            float rs = p0[r] + p1[r];
            rs += __shfl_xor(rs, 1, 32);
            rs += __shfl_xor(rs, 2, 32);
            rs += __shfl_xor(rs, 4, 32);
            rs += __shfl_xor(rs, 8, 32);
            l[r] = l[r] * corr[r] + rs;
            m[r] = mn;
        }

        // ---- rescale O, repack P through per-wave LDS slice -----------------
#pragma unroll
        for (int r = 0; r < 8; ++r) {
            O0[r] *= corr[r]; O1[r] *= corr[r]; O2[r] *= corr[r]; O3[r] *= corr[r];
            plds[wid][r + 8*g][n]      = (_Float16)p0[r];
            plds[wid][r + 8*g][16 + n] = (_Float16)p1[r];
        }
        __builtin_amdgcn_wave_barrier();
        const v8h* pr = (const v8h*)&plds[wid][n][0];
        const v16h pa = cat8(pr[0 + g], pr[2 + g]);
        __builtin_amdgcn_wave_barrier();

        // ---- O += P . V from LDS (rows 16t+n, 80-B padded stride) -----------
        const v8h* w0 = (const v8h*)&vst[cur][(unsigned)( 0 + n) * 40];
        const v8h* w1 = (const v8h*)&vst[cur][(unsigned)(16 + n) * 40];
        const v8h* w2 = (const v8h*)&vst[cur][(unsigned)(32 + n) * 40];
        const v8h* w3 = (const v8h*)&vst[cur][(unsigned)(48 + n) * 40];
        O0 = WMMA_F16(pa, cat8(w0[g], w0[2 + g]), O0);
        O1 = WMMA_F16(pa, cat8(w1[g], w1[2 + g]), O1);
        O2 = WMMA_F16(pa, cat8(w2[g], w2[2 + g]), O2);
        O3 = WMMA_F16(pa, cat8(w3[g], w3[2 + g]), O3);

        __syncthreads();   // all reads of buf[cur] done before it is re-filled
    }

    // ---- epilogue: O / l -> attn[b, n, h*64 + d] as f16
    const int b = bh / Hz, h = bh % Hz;
#pragma unroll
    for (int r = 0; r < 8; ++r) {
        const float inv = 1.0f / l[r];
        const int row   = mbase + r + 8*g;
        _Float16* orow = attn + ((size_t)b * Nz + row) * Cz + h * Dz;
        orow[n]      = (_Float16)(O0[r] * inv);
        orow[16 + n] = (_Float16)(O1[r] * inv);
        orow[32 + n] = (_Float16)(O2[r] * inv);
        orow[48 + n] = (_Float16)(O3[r] * inv);
    }
}

// ---------------------------------------------------------------------------
// Kernel 3: out = attn @ w_proj^T + b_proj  (same TDM weight staging)
// ---------------------------------------------------------------------------
__global__ __launch_bounds__(128)
void proj_kernel(const _Float16* __restrict__ attn, const float* __restrict__ w,
                 const float* __restrict__ bias, float* __restrict__ out)
{
    __shared__ alignas(16) float wstage[2][64 * 36];

    const int wid  = threadIdx.x >> 5;
    const int lane = threadIdx.x & 31;
    const int g    = lane >> 4;
    const int n    = lane & 15;

    const int tile  = blockIdx.x * 4 + wid;
    const int TM    = Mtot / 16;          // 1024 (all 4 waves share tf)
    const int tm    = tile % TM;
    const int tf    = tile / TM;          // 0..11
    const int mbase = tm * 16;
    const int obase = tf * 64;

    v8f acc[4] = {};
    const v8h*   arow  = (const v8h*)(attn + (size_t)(mbase + n) * Cz);
    const float* wbase = w + (size_t)obase * Cz;

    constexpr int NS = Cz / 32;   // 24
    RawH ha[2];
    ha[0] = load_rawh(arow, 0, g);
    if (wid == 0)
        tdm_load_2d(lds_off(&wstage[0][0]), wbase, 2, Cz, Cz, 32, 64, Cz, 4, 3);

#pragma unroll
    for (int s = 0; s < NS; ++s) {
        const int cur = s & 1, nxt = cur ^ 1;
        const int snext = (s + 1 < NS) ? (s + 1) : 0;   // wrap: last prefetch is dead
        ha[nxt] = load_rawh(arow, snext * 4, g);
        if (wid == 0) {
            tdm_load_2d(lds_off(&wstage[nxt][0]), wbase + snext * 32,
                        2, Cz, Cz, 32, 64, Cz, 4, 3);
            __builtin_amdgcn_s_wait_tensorcnt(1);       // tile s complete
        }
        __syncthreads();

        const v16h av = cat8(ha[cur].lo, ha[cur].hi);
#pragma unroll
        for (int t = 0; t < 4; ++t) {
            const float4* lrow = (const float4*)&wstage[cur][(16 * t + n) * 36];
            const v16h bv = cvt16(lrow[2*g], lrow[2*g + 1], lrow[4 + 2*g], lrow[5 + 2*g]);
            acc[t] = WMMA_F16(av, bv, acc[t]);
        }
        __syncthreads();
    }

#pragma unroll
    for (int t = 0; t < 4; ++t) {
        const int o    = obase + 16*t + n;
        const float bv = bias[o];
#pragma unroll
        for (int r = 0; r < 8; ++r) {
            const int tok = mbase + r + 8*g;
            out[(size_t)tok * Cz + o] = acc[t][r] + bv;
        }
    }
}

// ---------------------------------------------------------------------------
extern "C" void kernel_launch(void* const* d_in, const int* in_sizes, int n_in,
                              void* d_out, int out_size, void* d_ws, size_t ws_size,
                              hipStream_t stream)
{
    const float* x      = (const float*)d_in[0];
    const float* w_qkv  = (const float*)d_in[1];
    const float* w_proj = (const float*)d_in[2];
    const float* b_proj = (const float*)d_in[3];
    float* out = (float*)d_out;

    const size_t QSZ = (size_t)Bz * Hz * Nz * Dz;   // 12,582,912 halves per buffer
    _Float16* ws = (_Float16*)d_ws;
    _Float16* qb = ws;
    _Float16* kb = ws + QSZ;
    _Float16* vt = ws + 2 * QSZ;
    _Float16* at = ws + 3 * QSZ;

    qkv_kernel <<<(Mtot/16) * (F3/64) / 4, 128, 0, stream>>>(x, w_qkv, qb, kb, vt);
    attn_kernel<<<(Bz*Hz) * (Nz/16) / 4, 128, 0, stream>>>(qb, kb, vt, at);
    proj_kernel<<<(Mtot/16) * (Cz/64) / 4, 128, 0, stream>>>(at, w_proj, b_proj, out);
}
